// SDPFusion_35218731827336
// MI455X (gfx1250) — matrix operand order repaired and checked by reading, not compiled, not run on hardware
//
#include <hip/hip_runtime.h>
#include <hip/hip_bf16.h>

typedef __attribute__((ext_vector_type(16))) __bf16 v16bf;
typedef __attribute__((ext_vector_type(8)))  float  v8f;

#define C_DIM 256
#define HW_DIM 16384
#define W_DIM 128
#define YELEMS 33554432ull   // 8*256*16384 bf16 elements per tensor

// ---------------- WMMA fragment builders (contiguous k = 16h + s order, same for A and B) ----------------

// A-matrix 16x32 bf16: lanes 0-15 -> M=lane, K=0..15; lanes 16-31 -> M=lane-16, K=16..31
// (one 32B run per lane -> 2x ds_load_b128)
static __device__ __forceinline__ v16bf load_a_frag(const __bf16* base, int ld) {
    int lane = threadIdx.x & 31;
    int m  = lane & 15;
    int kb = (lane >> 4) << 4;          // 0 or 16
    const __bf16* p = base + m * ld;
    v16bf a;
#pragma unroll
    for (int i = 0; i < 16; ++i) {
        int k = kb + i;
        a[i] = p[k];
    }
    return a;
}

// B-matrix 32x16 bf16, source stored row=n, K-contiguous: lanes 0-15 -> N=lane, K=0..15;
// lanes 16-31 -> N=lane-16, K=16..31   (one 32B run per lane -> 2x ds_load_b128)
static __device__ __forceinline__ v16bf load_b_frag_kcontig(const __bf16* base, int ld) {
    int lane = threadIdx.x & 31;
    int n  = lane & 15;
    int kb = (lane >> 4) << 4;          // 0 or 16
    const __bf16* p = base + n * ld + kb;
    v16bf b;
#pragma unroll
    for (int i = 0; i < 16; ++i) b[i] = p[i];
    return b;
}

// ---------------- Kernel 1: 1x1-conv GEMM (bf16 WMMA) + GN partial stats ----------------
// grid: 2(t) * 8(b) * 2(mtile:128) * 256(ntile:64) = 8192 blocks of 256 threads (8 waves)
__global__ __launch_bounds__(256) void conv_gemm_kernel(
    const float* __restrict__ x_low, const float* __restrict__ x_high,
    const float* __restrict__ Wq, const float* __restrict__ Wk,
    __bf16* __restrict__ Yq, __bf16* __restrict__ Yk,
    float* __restrict__ partials)
{
    int wg = blockIdx.x;
    int nt = wg & 255;
    int mt = (wg >> 8) & 1;
    int b  = (wg >> 9) & 7;
    int t  = (wg >> 12) & 1;

    const float* X  = (t ? x_high : x_low) + (size_t)b * C_DIM * HW_DIM;
    const float* Wm = t ? Wk : Wq;
    __bf16*      Y  = (t ? Yk : Yq) + (size_t)b * C_DIM * HW_DIM;
    int mbase = mt * 128;
    int nbase = nt * 64;

    __shared__ __bf16 As[128][40];   // [m][k]   (80B row: 16B aligned)
    __shared__ __bf16 Bs[64][40];    // [n][k]   (transposed X tile)

    int tid  = threadIdx.x;
    int lane = tid & 31;
    int wid  = tid >> 5;

    // per-thread source positions for the cooperative tile loads
    int wr  = tid >> 1;               // W row 0..127
    int wc0 = (tid & 1) * 16;         // W col start
    int xk  = tid >> 3;               // X k-row 0..31
    int xj0 = (tid & 7) * 8;          // X col start

    v8f acc[4];
#pragma unroll
    for (int j = 0; j < 4; ++j)
#pragma unroll
        for (int i = 0; i < 8; ++i) acc[j][i] = 0.0f;

    for (int kk = 0; kk < 8; ++kk) {
        int k0 = kk * 32;
        {   // W tile: 128 rows x 32 cols, 16 consecutive floats per thread
            const float* src = Wm + (size_t)(mbase + wr) * C_DIM + k0 + wc0;
#pragma unroll
            for (int j = 0; j < 16; ++j) As[wr][wc0 + j] = (__bf16)src[j];
        }
        {   // X tile: 32 rows(k) x 64 cols(n), stored transposed [n][k]
            const float* src = X + (size_t)(k0 + xk) * HW_DIM + nbase + xj0;
#pragma unroll
            for (int j = 0; j < 8; ++j) Bs[xj0 + j][xk] = (__bf16)src[j];
        }
        if (kk < 7) {   // prefetch next K-step tiles (global_prefetch_b8) to hide HBM latency
            __builtin_prefetch(X + (size_t)(k0 + 32 + xk) * HW_DIM + nbase + xj0, 0, 1);
            __builtin_prefetch(Wm + (size_t)(mbase + wr) * C_DIM + k0 + 32 + wc0, 0, 1);
        }
        __syncthreads();
        v16bf a = load_a_frag(&As[wid * 16][0], 40);
#pragma unroll
        for (int j = 0; j < 4; ++j) {
            v16bf bf = load_b_frag_kcontig(&Bs[j * 16][0], 40);
            acc[j] = __builtin_amdgcn_wmma_f32_16x16x32_bf16(
                false, a, false, bf, (short)0, acc[j], false, false);
        }
        __syncthreads();
    }

    // store Y (bf16) + accumulate per-group-of-8-channels sum / sumsq
    int hi = lane >> 4;
    int ln = lane & 15;
    float s1 = 0.0f, s2 = 0.0f;
#pragma unroll
    for (int j = 0; j < 4; ++j) {
        int n = nbase + j * 16 + ln;
#pragma unroll
        for (int v = 0; v < 8; ++v) {
            int m = mbase + wid * 16 + hi * 8 + v;   // all 8 in one GN group
            float val = acc[j][v];
            Y[(size_t)m * HW_DIM + n] = (__bf16)val;
            s1 += val;
            s2 += val * val;
        }
    }
    // deterministic reduce across the 16 lanes of each half-wave
#pragma unroll
    for (int d = 1; d < 16; d <<= 1) {
        s1 += __shfl_xor(s1, d, 32);
        s2 += __shfl_xor(s2, d, 32);
    }
    if (ln == 0) {
        int g = mt * 16 + wid * 2 + hi;                 // group in [0,32)
        size_t pidx = ((((size_t)(t * 8 + b) * 32 + g) * 256) + nt) * 2;
        partials[pidx]     = s1;
        partials[pidx + 1] = s2;
    }
}

// ---------------- Kernel 2: fold partials -> {mean, rstd} (fixed order, deterministic) ----------------
__global__ __launch_bounds__(256) void gn_stats_kernel(
    const float* __restrict__ partials, float* __restrict__ stats)
{
    int idx = blockIdx.x * 256 + threadIdx.x;
    if (idx >= 512) return;                       // 2 tensors * 8 batches * 32 groups
    const float* p = partials + (size_t)idx * 512;
    float s1 = 0.0f, s2 = 0.0f;
    for (int i = 0; i < 256; ++i) { s1 += p[2 * i]; s2 += p[2 * i + 1]; }
    const float inv_n = 1.0f / 131072.0f;         // 8 channels * 16384 pixels
    float mu  = s1 * inv_n;
    float var = fmaxf(s2 * inv_n - mu * mu, 0.0f);
    stats[2 * idx]     = mu;
    stats[2 * idx + 1] = rsqrtf(var + 1e-5f);
}

// ---------------- Kernel 3: per-patch GN + attention (bf16 WMMA) + residual ----------------
// one block (4 waves) per patch; dynamic LDS layout:
//   kt  [64][264] bf16  (pos-major, B operand of S)
//   ktT [256][72] bf16  (ch-major, B operand of O = v^T view, built during load)
//   at  [64][72]  bf16  (attn weights, A operand of O)
//   region{ qt [64][264] bf16 + Sf [64][68] f32 }  reused as  Ot [256][66] f32
#define KT_LD    264
#define KTT_LD   72
#define AT_LD    72
#define SF_LD    68
#define OT_LD    66
#define KT_BYTES  (64 * KT_LD * 2)     // 33792
#define KTT_BYTES (256 * KTT_LD * 2)   // 36864
#define AT_BYTES  (64 * AT_LD * 2)     // 9216
#define RG_BYTES  (256 * OT_LD * 4)    // 67584 > qt(33792)+Sf(17408)
#define ATTN_LDS_BYTES (KT_BYTES + KTT_BYTES + AT_BYTES + RG_BYTES)   // 147456

__global__ __launch_bounds__(128) void attn_kernel(
    const __bf16* __restrict__ Yq, const __bf16* __restrict__ Yk,
    const float* __restrict__ stats,
    const float* __restrict__ gamq, const float* __restrict__ betq,
    const float* __restrict__ gamk, const float* __restrict__ betk,
    const float* __restrict__ x_low, float* __restrict__ out)
{
    int pid = blockIdx.x;
    int pw  = pid & 15;
    int ph  = (pid >> 4) & 15;
    int b   = pid >> 8;

    extern __shared__ __align__(16) char smem[];
    __bf16* kt  = (__bf16*)smem;
    __bf16* ktT = (__bf16*)(smem + KT_BYTES);
    __bf16* at  = (__bf16*)(smem + KT_BYTES + KTT_BYTES);
    char*   rg  = smem + KT_BYTES + KTT_BYTES + AT_BYTES;
    __bf16* qt  = (__bf16*)rg;
    float*  Sf  = (float*)(rg + 64 * KT_LD * 2);
    float*  Ot  = (float*)rg;

    __shared__ float aff[4][256];                // aq, bq, ak, bk per channel

    int tid  = threadIdx.x;
    int lane = tid & 31;
    int wid  = tid >> 5;

    // fold GN {mean,rstd}, gamma/beta, and q's 1/sqrt(C) scale into per-channel affine
    for (int c = tid; c < 256; c += 128) {
        int g = c >> 3;
        float muq = stats[(((0 * 8 + b) * 32 + g) << 1) + 0];
        float rsq = stats[(((0 * 8 + b) * 32 + g) << 1) + 1];
        float muk = stats[(((1 * 8 + b) * 32 + g) << 1) + 0];
        float rsk = stats[(((1 * 8 + b) * 32 + g) << 1) + 1];
        const float sc = 0.0625f;                // 1/sqrt(256)
        aff[0][c] = rsq * gamq[c] * sc;
        aff[1][c] = (betq[c] - muq * rsq * gamq[c]) * sc;
        aff[2][c] = rsk * gamk[c];
        aff[3][c] = betk[c] - muk * rsk * gamk[c];
    }
    __syncthreads();

    // load q/k patch tiles (8 contiguous bf16 = one 16B uint4 per (c,p1) row), apply affine;
    // k is written both pos-major (kt) and ch-major (ktT, one b128 store per row)
    size_t ybase = (size_t)b * C_DIM * HW_DIM + (size_t)(ph * 8) * W_DIM + (size_t)pw * 8;
    union U8 { uint4 u; __bf16 h[8]; };
    for (int it = 0; it < 16; ++it) {
        int idx = it * 128 + tid;                // 0..2047 = (c, p1)
        int c  = idx >> 3;
        int p1 = idx & 7;
        size_t ga = ybase + (size_t)c * HW_DIM + (size_t)p1 * W_DIM;
        U8 qv, kv, kn;
        qv.u = *(const uint4*)(Yq + ga);
        kv.u = *(const uint4*)(Yk + ga);
        float aq = aff[0][c], bq = aff[1][c], ak = aff[2][c], bk = aff[3][c];
#pragma unroll
        for (int j = 0; j < 8; ++j) {
            int pos = p1 * 8 + j;
            qt[pos * KT_LD + c] = (__bf16)((float)qv.h[j] * aq + bq);
            __bf16 kb = (__bf16)((float)kv.h[j] * ak + bk);
            kt[pos * KT_LD + c] = kb;
            kn.h[j] = kb;
        }
        *(uint4*)(ktT + c * KTT_LD + p1 * 8) = kn.u;   // 16B-aligned (144B row stride)
    }
    __syncthreads();

    // S = (q/16) . k : each wave owns 16 rows, 4x WMMA tiles across N, K=256
    int mb = wid * 16;
    v8f sacc[4];
#pragma unroll
    for (int j = 0; j < 4; ++j)
#pragma unroll
        for (int i = 0; i < 8; ++i) sacc[j][i] = 0.0f;
    for (int kk = 0; kk < 8; ++kk) {
        v16bf a = load_a_frag(qt + mb * KT_LD + kk * 32, KT_LD);
#pragma unroll
        for (int j = 0; j < 4; ++j) {
            v16bf bf = load_b_frag_kcontig(kt + (j * 16) * KT_LD + kk * 32, KT_LD);
            sacc[j] = __builtin_amdgcn_wmma_f32_16x16x32_bf16(
                false, a, false, bf, (short)0, sacc[j], false, false);
        }
    }
    int hi = lane >> 4, ln = lane & 15;
#pragma unroll
    for (int j = 0; j < 4; ++j)
#pragma unroll
        for (int v = 0; v < 8; ++v)
            Sf[(mb + hi * 8 + v) * SF_LD + j * 16 + ln] = sacc[j][v];
    __syncthreads();

    // row softmax -> attn (bf16); 2 threads per row, deterministic shfl-pair combine
    {
        int r = tid >> 1;                        // row 0..63
        int h = tid & 1;                         // column half
        float* row = Sf + r * SF_LD + h * 32;
        float mx = -1e30f;
#pragma unroll 8
        for (int c2 = 0; c2 < 32; ++c2) mx = fmaxf(mx, row[c2]);
        mx = fmaxf(mx, __shfl_xor(mx, 1, 32));
        float sum = 0.0f;
#pragma unroll 8
        for (int c2 = 0; c2 < 32; ++c2) { float e = __expf(row[c2] - mx); row[c2] = e; sum += e; }
        sum += __shfl_xor(sum, 1, 32);
        float inv = 1.0f / sum;
        __bf16* arow = at + r * AT_LD + h * 32;
#pragma unroll 8
        for (int c2 = 0; c2 < 32; ++c2) arow[c2] = (__bf16)(row[c2] * inv);
    }
    __syncthreads();   // qt/Sf dead past here -> region becomes Ot

    // O = attn . k^T : wave owns 16 rows, 16 WMMA tiles across 256 channels, K=64
    // B operand now comes K-contiguous from ktT -> 2x ds_load_b128 per fragment
    v16bf a0 = load_a_frag(at + mb * AT_LD + 0,  AT_LD);
    v16bf a1 = load_a_frag(at + mb * AT_LD + 32, AT_LD);
#pragma unroll
    for (int nt2 = 0; nt2 < 16; ++nt2) {
        v8f o;
#pragma unroll
        for (int i = 0; i < 8; ++i) o[i] = 0.0f;
        v16bf b0 = load_b_frag_kcontig(ktT + (nt2 * 16) * KTT_LD + 0,  KTT_LD);
        o = __builtin_amdgcn_wmma_f32_16x16x32_bf16(false, a0, false, b0, (short)0, o, false, false);
        v16bf b1 = load_b_frag_kcontig(ktT + (nt2 * 16) * KTT_LD + 32, KTT_LD);
        o = __builtin_amdgcn_wmma_f32_16x16x32_bf16(false, a1, false, b1, (short)0, o, false, false);
        int n = nt2 * 16 + ln;                   // channel
#pragma unroll
        for (int v = 0; v < 8; ++v)
            Ot[n * OT_LD + mb + hi * 8 + v] = o[v];   // transposed stage: [ch][pos]
    }
    __syncthreads();

    // writeout: out = O + x_low, coalesced along patch rows
    const float* xl = x_low + ybase;
    float*       op = out   + ybase;
    for (int it = 0; it < 128; ++it) {
        int idx = it * 128 + tid;                // 0..16383 = (c, pos)
        int c  = idx >> 6;
        int p  = idx & 63;
        size_t ga = (size_t)c * HW_DIM + (size_t)(p >> 3) * W_DIM + (p & 7);
        op[ga] = Ot[c * OT_LD + p] + xl[ga];
    }
}

// ---------------- host launch ----------------
extern "C" void kernel_launch(void* const* d_in, const int* in_sizes, int n_in,
                              void* d_out, int out_size, void* d_ws, size_t ws_size,
                              hipStream_t stream) {
    const float* x_low  = (const float*)d_in[0];
    const float* x_high = (const float*)d_in[1];
    const float* Wq     = (const float*)d_in[2];
    const float* gq     = (const float*)d_in[3];
    const float* bq     = (const float*)d_in[4];
    const float* Wk     = (const float*)d_in[5];
    const float* gk     = (const float*)d_in[6];
    const float* bk     = (const float*)d_in[7];
    float* out = (float*)d_out;

    // workspace: Yq | Yk (bf16, 64MiB each) | partials (1MiB) | stats (4KiB)
    __bf16* Yq = (__bf16*)d_ws;
    __bf16* Yk = Yq + YELEMS;
    float* partials = (float*)((char*)d_ws + 2ull * YELEMS * sizeof(__bf16));
    float* stats    = partials + 2ull * 8 * 32 * 256 * 2;

    (void)hipFuncSetAttribute((const void*)attn_kernel,
                              hipFuncAttributeMaxDynamicSharedMemorySize,
                              ATTN_LDS_BYTES);

    conv_gemm_kernel<<<dim3(8192), dim3(256), 0, stream>>>(
        x_low, x_high, Wq, Wk, Yq, Yk, partials);
    gn_stats_kernel<<<dim3(2), dim3(256), 0, stream>>>(partials, stats);
    attn_kernel<<<dim3(2048), dim3(128), ATTN_LDS_BYTES, stream>>>(
        Yq, Yk, stats, gq, bq, gk, bk, x_low, out);
}
